// MultiScaleMinGRULayer_19207093747865
// MI455X (gfx1250) — matrix operand order repaired and checked by reading, not compiled, not run on hardware
//
#include <hip/hip_runtime.h>

// MultiScaleMinGRU: B=4, S=8192, D=1024, N=4, K=256
//  z  = sigmoid(x @ Wz' + bz), ht = x @ Wh' + bh   (Wz'/Wh' are [D, N*K] views)
//  h  = scan(h_t = (1-z)h_{t-1} + z*ht) over S
//  out = sigmoid(h @ Wg + bg) * h

typedef __bf16 bf16_t;
typedef __attribute__((ext_vector_type(16))) __bf16 v16bf;
typedef __attribute__((ext_vector_type(8)))  __bf16 v8bf;
typedef __attribute__((ext_vector_type(8)))  float  v8f;

#define B_  4
#define S_  8192
#define D_  1024
#define N_  4
#define K_  256
#define M_  (B_ * S_)       // 32768 rows
#define CH  128             // chunks along S
#define CHLEN (S_ / CH)     // 64

#define TM 128
#define TK 32
#define NK (D_ / TK)        // 32 K-steps
#define LSTR 40             // LDS row stride: 40 elems = 80 B (16B-aligned rows, conflict-free)

__device__ __forceinline__ float sigmoidf_(float x) {
    // fast gate: v_rcp_f32 instead of the full IEEE divide sequence
    return __builtin_amdgcn_rcpf(1.0f + __expf(-x));
}

// Assemble a v16bf fragment from two aligned 16B LDS loads (pure register concat).
__device__ __forceinline__ v16bf frag16(const bf16_t* p_lo, const bf16_t* p_hi) {
    v8bf lo = *(const v8bf*)p_lo;
    v8bf hi = *(const v8bf*)p_hi;
    return __builtin_shufflevector(lo, hi, 0, 1, 2, 3, 4, 5, 6, 7,
                                           8, 9, 10, 11, 12, 13, 14, 15);
}

// CDNA5 async global->LDS copy, 16B per lane, tracked by ASYNCcnt.
__device__ __forceinline__ void async_ld16(unsigned lds_addr, unsigned long long gaddr) {
    asm volatile("global_load_async_to_lds_b128 %0, %1, off"
                 :: "v"(lds_addr), "v"(gaddr) : "memory");
}
__device__ __forceinline__ void wait_async0() {
    asm volatile("s_wait_asynccnt 0" ::: "memory");
}
__device__ __forceinline__ unsigned lds_addr_of(const void* p) {
    return (unsigned)(unsigned long long)p;   // LDS aperture: addr[31:0] is the LDS byte address
}

// ---------------- conversion kernels ----------------

__global__ __launch_bounds__(256) void cvt_x_kernel(const float4* __restrict__ x4,
                                                    bf16_t* __restrict__ xb, long n4) {
    long i = (long)blockIdx.x * 256 + threadIdx.x;
    if (i >= n4) return;
    float4 v = x4[i];
    bf16_t o[4] = {(bf16_t)v.x, (bf16_t)v.y, (bf16_t)v.z, (bf16_t)v.w};
    *(uint2*)(xb + i * 4) = *(const uint2*)o;
}

// Wz/Wh: [N, D, K] fp32 -> TRANSPOSED bf16 [N*K, D]  (row c = n*K+k, col d)
__global__ __launch_bounds__(256) void cvt_wzh_kernel(const float* __restrict__ Wz,
                                                      const float* __restrict__ Wh,
                                                      bf16_t* __restrict__ WzbT,
                                                      bf16_t* __restrict__ WhbT) {
    long i = (long)blockIdx.x * 256 + threadIdx.x;   // over N*D*K, source-coalesced
    if (i >= (long)N_ * D_ * K_) return;
    int k = (int)(i & (K_ - 1));
    long r = i >> 8;                 // n*D + d
    int d = (int)(r & (D_ - 1));
    int n = (int)(r >> 10);
    long dst = ((long)(n * K_ + k)) * D_ + d;
    WzbT[dst] = (bf16_t)Wz[i];
    WhbT[dst] = (bf16_t)Wh[i];
}

// Wg: [D, D] fp32 -> transposed bf16 [D, D] (row = out col, col = d)
__global__ __launch_bounds__(256) void cvt_wg_kernel(const float* __restrict__ Wg,
                                                     bf16_t* __restrict__ WgbT) {
    long i = (long)blockIdx.x * 256 + threadIdx.x;   // source-coalesced
    if (i >= (long)D_ * D_) return;
    int c = (int)(i & (D_ - 1));
    int d = (int)(i >> 10);
    WgbT[(long)c * D_ + d] = (bf16_t)Wg[i];
}

// ---------------- fused z / h~ GEMM (bf16 WMMA) ----------------
// Block tile: 128(M) x 64(N) per weight matrix, 8 waves, each wave 16x64 for
// BOTH Wz and Wh (A-tile reused). Double-buffered LDS: async DMA of tile k+1
// overlaps the 8 WMMAs on tile k; one barrier per K-step.

__global__ __launch_bounds__(256) void gemm_zh_kernel(const bf16_t* __restrict__ xb,
                                                      const bf16_t* __restrict__ WzbT,
                                                      const bf16_t* __restrict__ WhbT,
                                                      const float* __restrict__ bz,
                                                      const float* __restrict__ bh,
                                                      float* __restrict__ a_out,
                                                      float* __restrict__ b_out) {
    __shared__ bf16_t As[2][TM][LSTR];    // [buf][m][k]
    __shared__ bf16_t Bz[2][64][LSTR];    // [buf][n][k]  (K-contiguous)
    __shared__ bf16_t Bh[2][64][LSTR];

    const int tid  = threadIdx.x;
    const int lane = tid & 31;
    const int wave = tid >> 5;         // 0..7
    const int m0   = blockIdx.x * TM;
    const int c0   = blockIdx.y * 64;
    const int nlo  = lane & 15;
    const int hi   = lane >> 4;        // half-wave select

    // Async tile-fill assignments: 16B granules, 4 granules per 32-elem row.
    const int gr = tid >> 2;           // row 0..63
    const int gc = (tid & 3) * 8;      // elem 0,8,16,24
    unsigned long long gA0 = (unsigned long long)(xb   + (long)(m0 + gr)      * D_ + gc);
    unsigned long long gA1 = (unsigned long long)(xb   + (long)(m0 + gr + 64) * D_ + gc);
    unsigned long long gBz = (unsigned long long)(WzbT + (long)(c0 + gr)      * D_ + gc);
    unsigned long long gBh = (unsigned long long)(WhbT + (long)(c0 + gr)      * D_ + gc);
    unsigned lA0[2], lA1[2], lBz[2], lBh[2];
#pragma unroll
    for (int b = 0; b < 2; ++b) {
        lA0[b] = lds_addr_of(&As[b][gr][gc]);
        lA1[b] = lds_addr_of(&As[b][gr + 64][gc]);
        lBz[b] = lds_addr_of(&Bz[b][gr][gc]);
        lBh[b] = lds_addr_of(&Bh[b][gr][gc]);
    }

    v8f accz[4], acch[4];
#pragma unroll
    for (int j = 0; j < 4; ++j)
#pragma unroll
        for (int r = 0; r < 8; ++r) { accz[j][r] = 0.0f; acch[j][r] = 0.0f; }

    // prologue: DMA tile 0 into buffer 0
    async_ld16(lA0[0], gA0);
    async_ld16(lA1[0], gA1);
    async_ld16(lBz[0], gBz);
    async_ld16(lBh[0], gBh);
    gA0 += TK * 2; gA1 += TK * 2; gBz += TK * 2; gBh += TK * 2;

#pragma unroll 2
    for (int it = 0; it < NK; ++it) {
        const int buf = it & 1;
        wait_async0();                 // this wave's portion of tile `it` landed
        __syncthreads();               // all waves landed tile `it`; all done reading buf^1
        if (it + 1 < NK) {             // DMA tile it+1 overlaps compute of tile it
            async_ld16(lA0[buf ^ 1], gA0);
            async_ld16(lA1[buf ^ 1], gA1);
            async_ld16(lBz[buf ^ 1], gBz);
            async_ld16(lBh[buf ^ 1], gBh);
            gA0 += TK * 2; gA1 += TK * 2; gBz += TK * 2; gBh += TK * 2;
        }

        // A fragment: lane L = row (wave*16 + L%16); two contiguous 16B runs.
        const bf16_t* arow = &As[buf][wave * 16 + nlo][0];
        v16bf af = frag16(&arow[hi * 16], &arow[hi * 16 + 8]);

#pragma unroll
        for (int j = 0; j < 4; ++j) {
            const bf16_t* bzr = &Bz[buf][j * 16 + nlo][hi * 16];
            const bf16_t* bhr = &Bh[buf][j * 16 + nlo][hi * 16];
            v16bf bzf = frag16(bzr, bzr + 8);
            v16bf bhf = frag16(bhr, bhr + 8);
            accz[j] = __builtin_amdgcn_wmma_f32_16x16x32_bf16(
                false, af, false, bzf, (short)0, accz[j], false, false);
            acch[j] = __builtin_amdgcn_wmma_f32_16x16x32_bf16(
                false, af, false, bhf, (short)0, acch[j], false, false);
        }
    }

    // Epilogue: C/D layout — VGPR r holds row (r + hi*8), col = nlo
#pragma unroll
    for (int j = 0; j < 4; ++j) {
#pragma unroll
        for (int r = 0; r < 8; ++r) {
            int row = m0 + wave * 16 + hi * 8 + r;
            int col = c0 + j * 16 + nlo;
            long off = (long)row * D_ + col;
            float z  = sigmoidf_(accz[j][r] + bz[col]);
            float hp = acch[j][r] + bh[col];
            a_out[off] = 1.0f - z;
            b_out[off] = z * hp;
        }
    }
}

// ---------------- chunked associative scan over S ----------------

// pass 1: per (b, d, chunk) aggregate (A = prod a, B = fold b)
__global__ __launch_bounds__(256) void scan_pass1(const float* __restrict__ a_buf,
                                                  const float* __restrict__ b_buf,
                                                  float* __restrict__ Ag,
                                                  float* __restrict__ Bg) {
    long t = (long)blockIdx.x * 256 + threadIdx.x;
    if (t >= (long)B_ * D_ * CH) return;
    int d   = (int)(t & (D_ - 1));
    long r  = t >> 10;                // over B*CH
    int bi  = (int)(r >> 7);          // /CH
    int ch  = (int)(r & (CH - 1));
    const long base = ((long)bi * S_ + (long)ch * CHLEN) * D_ + d;
    float A = 1.0f, Bv = 0.0f;
    for (int s = 0; s < CHLEN; ++s) {
        float a = a_buf[base + (long)s * D_];
        float b = b_buf[base + (long)s * D_];
        Bv = a * Bv + b;
        A *= a;
    }
    long gi = ((long)bi * D_ + d) * CH + ch;
    Ag[gi] = A;
    Bg[gi] = Bv;
}

// pass 2: serial scan across 128 chunk aggregates per (b, d) -> per-chunk h_init
__global__ __launch_bounds__(256) void scan_pass2(const float* __restrict__ Ag,
                                                  const float* __restrict__ Bg,
                                                  float* __restrict__ Hinit) {
    int t = blockIdx.x * 256 + threadIdx.x;
    if (t >= B_ * D_) return;
    long base = (long)t * CH;
    float h = 0.0f;
    for (int ch = 0; ch < CH; ++ch) {
        Hinit[base + ch] = h;
        h = Ag[base + ch] * h + Bg[base + ch];
    }
}

// pass 3: replay chunks from h_init, emit h (f32) and h (bf16)
__global__ __launch_bounds__(256) void scan_pass3(const float* __restrict__ a_buf,
                                                  const float* __restrict__ b_buf,
                                                  const float* __restrict__ Hinit,
                                                  float* __restrict__ h_buf,
                                                  bf16_t* __restrict__ hb) {
    long t = (long)blockIdx.x * 256 + threadIdx.x;
    if (t >= (long)B_ * D_ * CH) return;
    int d  = (int)(t & (D_ - 1));
    long r = t >> 10;
    int bi = (int)(r >> 7);
    int ch = (int)(r & (CH - 1));
    const long base = ((long)bi * S_ + (long)ch * CHLEN) * D_ + d;
    float h = Hinit[((long)bi * D_ + d) * CH + ch];
    for (int s = 0; s < CHLEN; ++s) {
        long off = base + (long)s * D_;
        h = a_buf[off] * h + b_buf[off];
        h_buf[off] = h;
        hb[off]    = (bf16_t)h;
    }
}

// ---------------- output gate GEMM (bf16 WMMA) ----------------
// Block tile 128x128, 8 waves, each wave 16x128, double-buffered async DMA.
// out = sigmoid(h@Wg + bg) * h.

__global__ __launch_bounds__(256) void gemm_out_kernel(const bf16_t* __restrict__ hbm,
                                                       const bf16_t* __restrict__ WgbT,
                                                       const float* __restrict__ bg,
                                                       const float* __restrict__ h_buf,
                                                       float* __restrict__ out) {
    __shared__ bf16_t As[2][TM][LSTR];    // [buf][m][k]
    __shared__ bf16_t Bt[2][128][LSTR];   // [buf][n][k]  (K-contiguous)

    const int tid  = threadIdx.x;
    const int lane = tid & 31;
    const int wave = tid >> 5;
    const int m0   = blockIdx.x * TM;
    const int c0   = blockIdx.y * 128;
    const int nlo  = lane & 15;
    const int hi   = lane >> 4;

    const int gr = tid >> 2;
    const int gc = (tid & 3) * 8;
    unsigned long long gA0 = (unsigned long long)(hbm  + (long)(m0 + gr)      * D_ + gc);
    unsigned long long gA1 = (unsigned long long)(hbm  + (long)(m0 + gr + 64) * D_ + gc);
    unsigned long long gB0 = (unsigned long long)(WgbT + (long)(c0 + gr)      * D_ + gc);
    unsigned long long gB1 = (unsigned long long)(WgbT + (long)(c0 + gr + 64) * D_ + gc);
    unsigned lA0[2], lA1[2], lB0[2], lB1[2];
#pragma unroll
    for (int b = 0; b < 2; ++b) {
        lA0[b] = lds_addr_of(&As[b][gr][gc]);
        lA1[b] = lds_addr_of(&As[b][gr + 64][gc]);
        lB0[b] = lds_addr_of(&Bt[b][gr][gc]);
        lB1[b] = lds_addr_of(&Bt[b][gr + 64][gc]);
    }

    v8f acc[8];
#pragma unroll
    for (int j = 0; j < 8; ++j)
#pragma unroll
        for (int r = 0; r < 8; ++r) acc[j][r] = 0.0f;

    async_ld16(lA0[0], gA0);
    async_ld16(lA1[0], gA1);
    async_ld16(lB0[0], gB0);
    async_ld16(lB1[0], gB1);
    gA0 += TK * 2; gA1 += TK * 2; gB0 += TK * 2; gB1 += TK * 2;

#pragma unroll 2
    for (int it = 0; it < NK; ++it) {
        const int buf = it & 1;
        wait_async0();
        __syncthreads();
        if (it + 1 < NK) {
            async_ld16(lA0[buf ^ 1], gA0);
            async_ld16(lA1[buf ^ 1], gA1);
            async_ld16(lB0[buf ^ 1], gB0);
            async_ld16(lB1[buf ^ 1], gB1);
            gA0 += TK * 2; gA1 += TK * 2; gB0 += TK * 2; gB1 += TK * 2;
        }

        const bf16_t* arow = &As[buf][wave * 16 + nlo][0];
        v16bf af = frag16(&arow[hi * 16], &arow[hi * 16 + 8]);

#pragma unroll
        for (int j = 0; j < 8; ++j) {
            const bf16_t* brow = &Bt[buf][j * 16 + nlo][hi * 16];
            v16bf bf = frag16(brow, brow + 8);
            acc[j] = __builtin_amdgcn_wmma_f32_16x16x32_bf16(
                false, af, false, bf, (short)0, acc[j], false, false);
        }
    }

#pragma unroll
    for (int j = 0; j < 8; ++j) {
#pragma unroll
        for (int r = 0; r < 8; ++r) {
            int row = m0 + wave * 16 + hi * 8 + r;
            int col = c0 + j * 16 + nlo;
            long off = (long)row * D_ + col;
            float g = sigmoidf_(acc[j][r] + bg[col]);
            out[off] = g * h_buf[off];
        }
    }
}

// ---------------- launcher ----------------

extern "C" void kernel_launch(void* const* d_in, const int* in_sizes, int n_in,
                              void* d_out, int out_size, void* d_ws, size_t ws_size,
                              hipStream_t stream) {
    const float* x  = (const float*)d_in[0];   // [B,S,D]
    const float* Wz = (const float*)d_in[1];   // [N,D,K]
    const float* bz = (const float*)d_in[2];   // [N,K] flat = [1024]
    const float* Wh = (const float*)d_in[3];   // [N,D,K]
    const float* bh = (const float*)d_in[4];   // [N,K]
    const float* Wg = (const float*)d_in[5];   // [D,D]
    const float* bg = (const float*)d_in[6];   // [D]
    float* out = (float*)d_out;

    // workspace carve-out (256B aligned)
    char* ws = (char*)d_ws;
    size_t off = 0;
    auto carve = [&](size_t bytes) -> char* {
        char* p = ws + off;
        off = (off + bytes + 255) & ~(size_t)255;
        return p;
    };
    const long MD = (long)M_ * D_;                   // 33,554,432
    bf16_t* xb    = (bf16_t*)carve(MD * 2);
    bf16_t* WzbT  = (bf16_t*)carve((size_t)D_ * D_ * 2);
    bf16_t* WhbT  = (bf16_t*)carve((size_t)D_ * D_ * 2);
    bf16_t* WgbT  = (bf16_t*)carve((size_t)D_ * D_ * 2);
    float*  a_buf = (float*)carve(MD * 4);
    float*  b_buf = (float*)carve(MD * 4);
    float*  h_buf = (float*)carve(MD * 4);
    bf16_t* hb    = (bf16_t*)carve(MD * 2);
    float*  Ag    = (float*)carve((size_t)B_ * D_ * CH * 4);
    float*  Bg    = (float*)carve((size_t)B_ * D_ * CH * 4);
    float*  Hinit = (float*)carve((size_t)B_ * D_ * CH * 4);
    (void)ws_size; (void)in_sizes; (void)n_in; (void)out_size;

    // 1) precision conversion (+ weight transpose for K-contiguous B tiles)
    cvt_x_kernel<<<(unsigned)(MD / 4 / 256), 256, 0, stream>>>((const float4*)x, xb, MD / 4);
    cvt_wzh_kernel<<<(D_ * D_) / 256, 256, 0, stream>>>(Wz, Wh, WzbT, WhbT);
    cvt_wg_kernel<<<(D_ * D_) / 256, 256, 0, stream>>>(Wg, WgbT);

    // 2) fused z / h~ GEMMs -> a = 1-z, b = z*h~
    dim3 gzh(M_ / TM, D_ / 64);
    gemm_zh_kernel<<<gzh, 256, 0, stream>>>(xb, WzbT, WhbT, bz, bh, a_buf, b_buf);

    // 3) chunked scan over S
    long nThreads1 = (long)B_ * D_ * CH;
    scan_pass1<<<(unsigned)(nThreads1 / 256), 256, 0, stream>>>(a_buf, b_buf, Ag, Bg);
    scan_pass2<<<(B_ * D_) / 256, 256, 0, stream>>>(Ag, Bg, Hinit);
    scan_pass3<<<(unsigned)(nThreads1 / 256), 256, 0, stream>>>(a_buf, b_buf, Hinit, h_buf, hb);

    // 4) output gate GEMM with fused sigmoid*h
    gemm_out_kernel<<<dim3(M_ / TM, D_ / 128), 256, 0, stream>>>(hb, WgbT, bg, h_buf, out);
}
